// GCNAutoEncoder_89300960019182
// MI455X (gfx1250) — matrix operand order repaired and checked by reading, not compiled, not run on hardware
//
#include <hip/hip_runtime.h>
#include <hip/hip_bf16.h>

typedef __attribute__((ext_vector_type(16))) __bf16 v16bf;
typedef __attribute__((ext_vector_type(4)))  __bf16 v4bf;
typedef __attribute__((ext_vector_type(8)))  float  v8f;

__device__ __forceinline__ void atomAddF(float* p, float v) {
    unsafeAtomicAdd(p, v);      // hardware global_atomic_add_f32 (no-return)
}

// ---------------- degree / normalization ----------------
__global__ void k_fill(float* __restrict__ p, float v, int n) {
    int i = blockIdx.x * blockDim.x + threadIdx.x;
    if (i < n) p[i] = v;
}
__global__ void k_count_deg(const int* __restrict__ dst, float* __restrict__ deg, int E) {
    int e = blockIdx.x * blockDim.x + threadIdx.x;
    if (e < E) atomAddF(&deg[dst[e]], 1.0f);
}
__global__ void k_rsqrt(float* __restrict__ d, int n) {
    int i = blockIdx.x * blockDim.x + threadIdx.x;
    if (i < n) d[i] = rsqrtf(d[i]);     // deg >= 1 always (self-loop)
}
__global__ void k_enorm(const int* __restrict__ src, const int* __restrict__ dst,
                        const float* __restrict__ dinv, float* __restrict__ enorm, int E) {
    int e = blockIdx.x * blockDim.x + threadIdx.x;
    if (e < E) enorm[e] = dinv[src[e]] * dinv[dst[e]];
}

// ------------- one-shot conversions (hoisted out of GEMM inner loop) -------------
template<bool RELU>
__global__ void k_cvt_bf16(const float* __restrict__ in, __bf16* __restrict__ out, int n) {
    int i = 4 * (blockIdx.x * blockDim.x + threadIdx.x);
    if (i + 3 < n) {
        float4 v = *(const float4*)(in + i);
        if (RELU) {
            v.x = fmaxf(v.x, 0.f); v.y = fmaxf(v.y, 0.f);
            v.z = fmaxf(v.z, 0.f); v.w = fmaxf(v.w, 0.f);
        }
        *(v4bf*)(out + i) = (v4bf){(__bf16)v.x, (__bf16)v.y, (__bf16)v.z, (__bf16)v.w};
    } else {
        for (; i < n; ++i) {
            float v = in[i];
            if (RELU) v = fmaxf(v, 0.f);
            out[i] = (__bf16)v;
        }
    }
}
// weights: f32 [K,F] -> bf16 transposed [F,K] so WMMA B fragments are contiguous
__global__ void k_pack_wt(const float* __restrict__ W, __bf16* __restrict__ Wt,
                          int K, int F) {
    int idx = blockIdx.x * blockDim.x + threadIdx.x;
    if (idx >= K * F) return;
    int c = idx / K, k = idx - c * K;
    Wt[idx] = (__bf16)W[(size_t)k * F + c];
}

// ---- WMMA GEMM + fused self-loop epilogue:
//   P[n,c] = (Xb @ Wt^T)[n,c]
//   Y[n,c] = bias[c] + dinv[n]^2 * P[n,c]      (self-loop term of the aggregation)
template<int FOUT>
__global__ void k_gemm_wmma(const __bf16* __restrict__ Xb,   // [N,128] bf16 row-major
                            const __bf16* __restrict__ Wt,   // [FOUT,128] bf16
                            const float* __restrict__ bias,  // [FOUT]
                            const float* __restrict__ dinv,  // [N]
                            float* __restrict__ P,
                            float* __restrict__ Y, int N) {
    constexpr int K = 128;
    constexpr int WPB = FOUT / 16;              // waves per block
    __shared__ __bf16 sx[16 * K];               // 4KB bf16 row tile

    const int m0 = blockIdx.x * 16;
    const int nthreads = WPB * 32;
    const uint4* gsrc = (const uint4*)(Xb + (size_t)m0 * K);   // contiguous tile
    for (int i = threadIdx.x; i < (16 * K) / 8; i += nthreads)
        ((uint4*)sx)[i] = gsrc[i];
    __syncthreads();

    const int wave = threadIdx.x >> 5;
    const int lane = threadIdx.x & 31;
    const int g = lane >> 4;                    // half-wave group
    const int r = lane & 15;
    const int c0 = wave * 16;

    // Preload all 4 B fragments: Wt row (c0+r), K = 32*kb + 16g + [0,16)
    union { v16bf v; uint4 u[2]; } b[4];
    const uint4* wrow = (const uint4*)(Wt + (size_t)(c0 + r) * K);
    #pragma unroll
    for (int kb = 0; kb < 4; ++kb) {
        b[kb].u[0] = wrow[kb * 4 + 2 * g];
        b[kb].u[1] = wrow[kb * 4 + 2 * g + 1];
    }

    // K loop: two ds_load_b128 per step build the A fragment
    const uint4* srow = (const uint4*)sx;       // 8 bf16 per chunk
    v8f acc = {};
    #pragma unroll
    for (int kb = 0; kb < 4; ++kb) {
        union { v16bf v; uint4 u[2]; } a;
        a.u[0] = srow[r * 16 + kb * 4 + 2 * g];     // K = 32kb + 16g + [0,8)
        a.u[1] = srow[r * 16 + kb * 4 + 2 * g + 1]; // K = 32kb + 16g + 8 + [0,8)
        acc = __builtin_amdgcn_wmma_f32_16x16x32_bf16(
                  false, a.v, false, b[kb].v, (short)0, acc, false, false);
    }

    // Fused epilogue. C/D layout: row = m0 + v + 8*(lane/16), col = c0 + r.
    const float bcol = bias[c0 + r];
    if (m0 + 16 <= N) {                         // uniform branch: full tile
        #pragma unroll
        for (int v = 0; v < 8; ++v) {
            const int row = m0 + v + 8 * g;
            const size_t o = (size_t)row * FOUT + c0 + r;
            const float di = dinv[row];
            P[o] = acc[v];
            Y[o] = bcol + di * di * acc[v];
        }
    } else {                                    // tail tile only
        #pragma unroll
        for (int v = 0; v < 8; ++v) {
            const int row = m0 + v + 8 * g;
            if (row < N) {
                const size_t o = (size_t)row * FOUT + c0 + r;
                const float di = dinv[row];
                P[o] = acc[v];
                Y[o] = bcol + di * di * acc[v];
            }
        }
    }
}

// -------- aggregation: Y[dst] += enorm * P[src]; one WAVE per edge --------
// Edge scalars become s_loads (uniform via readfirstlane); payload is a
// coalesced vector gather from the L2-resident P buffer + fp32 atomics.
template<int F>
__global__ void k_edge_agg(const int* __restrict__ src, const int* __restrict__ dst,
                           const float* __restrict__ en, const float* __restrict__ P,
                           float* __restrict__ Y, int E) {
    constexpr int C = F / 32;                   // floats per lane
    int wid = blockIdx.x * (blockDim.x >> 5) + (threadIdx.x >> 5);
    int e = __builtin_amdgcn_readfirstlane(wid);
    if (e >= E) return;
    int s = src[e], d = dst[e];
    float w = en[e];
    int c = (threadIdx.x & 31) * C;
    const float* p = P + (size_t)s * F + c;
    float* y = Y + (size_t)d * F + c;
    float pv[C];
    #pragma unroll
    for (int j = 0; j < C; ++j) pv[j] = p[j];   // merged to b64/b128 load
    #pragma unroll
    for (int j = 0; j < C; ++j) atomAddF(y + j, w * pv[j]);
}

// ---------------- bottleneck: ZB = LayerNorm(relu(Z) @ Wm + bm) ----------------
__global__ void k_bottleneck(const float* __restrict__ Z,   // [N,64]
                             const float* __restrict__ Wm,  // [64,5]
                             const float* __restrict__ bm,
                             const float* __restrict__ lnw, const float* __restrict__ lnb,
                             float* __restrict__ ZB, int N) {
    __shared__ float sW[64 * 5];
    __shared__ float sb[5], slw[5], slb[5];
    for (int i = threadIdx.x; i < 320; i += blockDim.x) sW[i] = Wm[i];
    if (threadIdx.x < 5) {
        sb[threadIdx.x]  = bm[threadIdx.x];
        slw[threadIdx.x] = lnw[threadIdx.x];
        slb[threadIdx.x] = lnb[threadIdx.x];
    }
    __syncthreads();
    int n = blockIdx.x * blockDim.x + threadIdx.x;
    if (n >= N) return;
    float acc[5];
    #pragma unroll
    for (int j = 0; j < 5; ++j) acc[j] = sb[j];
    const float* zr = Z + (size_t)n * 64;
    #pragma unroll 8
    for (int k = 0; k < 64; ++k) {
        float v = fmaxf(zr[k], 0.f);
        #pragma unroll
        for (int j = 0; j < 5; ++j) acc[j] += v * sW[k * 5 + j];
    }
    float mu = 0.f;
    #pragma unroll
    for (int j = 0; j < 5; ++j) mu += acc[j];
    mu *= 0.2f;
    float var = 0.f;
    #pragma unroll
    for (int j = 0; j < 5; ++j) { float dd = acc[j] - mu; var += dd * dd; }
    var *= 0.2f;
    float is = rsqrtf(var + 1e-5f);
    #pragma unroll
    for (int j = 0; j < 5; ++j)
        ZB[(size_t)n * 5 + j] = (acc[j] - mu) * is * slw[j] + slb[j];
}

// ------- decoder layer 1 GEMM (K=5) + fused self-loop epilogue -------
__global__ void k_gemm_k5(const float* __restrict__ ZB,    // [N,5]
                          const float* __restrict__ W,     // [5,128]
                          const float* __restrict__ bias,  // [128]
                          const float* __restrict__ dinv,  // [N]
                          float* __restrict__ P,
                          float* __restrict__ Y, int N) {
    __shared__ float sW[5 * 128];
    for (int i = threadIdx.x; i < 640; i += blockDim.x) sW[i] = W[i];
    __syncthreads();
    int idx = blockIdx.x * blockDim.x + threadIdx.x;
    if (idx >= N * 128) return;
    int n = idx >> 7, f = idx & 127;
    const float* z = ZB + (size_t)n * 5;
    float acc = z[0] * sW[f] + z[1] * sW[128 + f] + z[2] * sW[256 + f]
              + z[3] * sW[384 + f] + z[4] * sW[512 + f];
    float di = dinv[n];
    P[idx] = acc;
    Y[idx] = bias[f] + di * di * acc;
}

// ---------------- host orchestration ----------------
extern "C" void kernel_launch(void* const* d_in, const int* in_sizes, int n_in,
                              void* d_out, int out_size, void* d_ws, size_t ws_size,
                              hipStream_t stream) {
    const float* x   = (const float*)d_in[0];
    const int*   ei  = (const int*)  d_in[1];
    const float* W1e = (const float*)d_in[2];  const float* b1e = (const float*)d_in[3];
    const float* W2e = (const float*)d_in[4];  const float* b2e = (const float*)d_in[5];
    const float* Wm  = (const float*)d_in[6];  const float* bm  = (const float*)d_in[7];
    const float* lnw = (const float*)d_in[8];  const float* lnb = (const float*)d_in[9];
    const float* W1d = (const float*)d_in[10]; const float* b1d = (const float*)d_in[11];
    const float* W2d = (const float*)d_in[12]; const float* b2d = (const float*)d_in[13];

    const int N = in_sizes[0] / 128;
    const int E = in_sizes[1] / 2;
    const int* src = ei;
    const int* dst = ei + E;

    float* ws    = (float*)d_ws;
    float* dinv  = ws;  ws += N;
    float* enorm = ws;  ws += E;
    float* P     = ws;  ws += (size_t)N * 128;   // GEMM output (pre-aggregation)
    float* Y     = ws;  ws += (size_t)N * 128;   // aggregated features
    float* ZB    = ws;  ws += (size_t)N * 5;     // bottleneck code
    __bf16* Xb   = (__bf16*)ws;  ws += (size_t)N * 64;   // bf16 activations (N*128)
    __bf16* Wt1  = (__bf16*)ws;  ws += 128 * 128 / 2;    // packed W1e^T
    __bf16* Wt2  = (__bf16*)ws;  ws += 128 * 64 / 2;     // packed W2e^T
    __bf16* Wt4  = (__bf16*)ws;  ws += 128 * 128 / 2;    // packed W2d^T
    float* OUT   = (float*)d_out;

    auto cdiv = [](long a, long b) { return (int)((a + b - 1) / b); };
    const int NT = cdiv(N, 16);                  // 16-row WMMA tiles
    const long NF = (long)N * 128;

    // degrees + per-edge norms
    k_fill     <<<cdiv(N, 256), 256, 0, stream>>>(dinv, 1.0f, N);   // self-loop deg
    k_count_deg<<<cdiv(E, 256), 256, 0, stream>>>(dst, dinv, E);
    k_rsqrt    <<<cdiv(N, 256), 256, 0, stream>>>(dinv, N);
    k_enorm    <<<cdiv(E, 256), 256, 0, stream>>>(src, dst, dinv, enorm, E);

    // pack weights once: transpose + bf16
    k_pack_wt<<<cdiv(128 * 128, 256), 256, 0, stream>>>(W1e, Wt1, 128, 128);
    k_pack_wt<<<cdiv(128 * 64,  256), 256, 0, stream>>>(W2e, Wt2, 128, 64);
    k_pack_wt<<<cdiv(128 * 128, 256), 256, 0, stream>>>(W2d, Wt4, 128, 128);

    // encoder GCN1: h = relu(agg(x @ W1e) + b1e)   [F=128]
    k_cvt_bf16<false><<<cdiv(NF, 1024), 256, 0, stream>>>(x, Xb, (int)NF);
    k_gemm_wmma<128><<<NT, 256, 0, stream>>>(Xb, Wt1, b1e, dinv, P, Y, N);
    k_edge_agg <128><<<cdiv(E, 8), 256, 0, stream>>>(src, dst, enorm, P, Y, E);

    // encoder GCN2: z = agg(relu(h) @ W2e) + b2e   [F=64]  (ReLU fused into cvt)
    k_cvt_bf16<true><<<cdiv(NF, 1024), 256, 0, stream>>>(Y, Xb, (int)NF);
    k_gemm_wmma<64><<<NT, 128, 0, stream>>>(Xb, Wt2, b2e, dinv, P, Y, N);
    k_edge_agg <64><<<cdiv(E, 8), 256, 0, stream>>>(src, dst, enorm, P, Y, E);

    // bottleneck: ZB = LN(relu(z) @ Wm + bm)
    k_bottleneck<<<cdiv(N, 256), 256, 0, stream>>>(Y, Wm, bm, lnw, lnb, ZB, N);

    // decoder GCN1: d = relu(agg(ZB @ W1d) + b1d)  [F=128, K=5]
    k_gemm_k5<<<cdiv(NF, 256), 256, 0, stream>>>(ZB, W1d, b1d, dinv, P, Y, N);
    k_edge_agg <128><<<cdiv(E, 8), 256, 0, stream>>>(src, dst, enorm, P, Y, E);

    // decoder GCN2: out = agg(relu(d) @ W2d) + b2d [F=128]  (ReLU fused into cvt)
    k_cvt_bf16<true><<<cdiv(NF, 1024), 256, 0, stream>>>(Y, Xb, (int)NF);
    k_gemm_wmma<128><<<NT, 256, 0, stream>>>(Xb, Wt4, b2d, dinv, P, OUT, N);
    k_edge_agg <128><<<cdiv(E, 8), 256, 0, stream>>>(src, dst, enorm, P, OUT, E);
}
